// FFNNLM_90434831385111
// MI455X (gfx1250) — matrix operand order repaired and checked by reading, not compiled, not run on hardware
//
#include <hip/hip_runtime.h>
#include <hip/hip_bf16.h>

typedef _Float16 half16 __attribute__((ext_vector_type(16)));
typedef _Float16 half8  __attribute__((ext_vector_type(8)));
typedef float    float8 __attribute__((ext_vector_type(8)));

#define TOKENS 8192   // B*S = 16*512
#define EMB    256
#define HID    1024
#define VOCAB  32000
#define VPAD   32768  // vocab padded to multiple of 4096
#define NCHUNK 8      // VPAD / 4096

// ---------- WMMA helpers (gfx1250 v_wmma_f32_16x16x32_f16) ----------

__device__ inline float8 wmma_f16(half16 a, half16 b, float8 c) {
  // 8 args: (neg_a, A, neg_b, B, c_mod, C, reuse_a, reuse_b)
  return __builtin_amdgcn_wmma_f32_16x16x32_f16(false, a, false, b, (short)0, c,
                                                false, false);
}

// A fragment (16x32, M x K), row-major source, ld in elements.
// Contiguous k order used here: lanes 0-15 hold M=lane, K in [0..15];
//             lanes 16-31 hold M=lane-16, K in [16..31].
__device__ inline half16 load_a_frag(const _Float16* A, int ld, int m0, int k0) {
  int lane = threadIdx.x & 31;
  const _Float16* p = A + (size_t)(m0 + (lane & 15)) * ld + k0 + ((lane >> 4) * 16);
  half8 lo = *(const half8*)p;
  half8 hi = *(const half8*)(p + 8);
  half16 r;
#pragma unroll
  for (int i = 0; i < 8; ++i) { r[i] = lo[i]; r[i + 8] = hi[i]; }
  return r;
}

// B fragment (32x16, K x N) from transposed storage Bt[N][K] (ldb = K).
// Lanes 0-15: col = lane, K=0..15; lanes 16-31: col = lane-16, K=16..31.
__device__ inline half16 load_b_frag(const _Float16* Bt, int ldb, int n0, int k0) {
  int lane = threadIdx.x & 31;
  const _Float16* p = Bt + (size_t)(n0 + (lane & 15)) * ldb + k0 + ((lane >> 4) * 16);
  return *(const half16*)p;
}

// ---------- Prep kernels ----------

// Tiled transpose-convert: out[n*K + k] = (f16) W[k*N + n]; zero-fill n >= N.
// Block (32,8), 32x32 tile via LDS -> both read and write sides coalesced.
__global__ void conv_transpose_f16(const float* __restrict__ W,
                                   _Float16* __restrict__ out,
                                   int K, int N, int Npad) {
  __shared__ float tile[32][33];
  int n0 = blockIdx.x * 32;
  int k0 = blockIdx.y * 32;
  int tx = threadIdx.x, ty = threadIdx.y;
#pragma unroll
  for (int i = 0; i < 32; i += 8) {
    int k = k0 + ty + i, n = n0 + tx;
    tile[ty + i][tx] = (n < N) ? W[(size_t)k * N + n] : 0.0f;
  }
  __syncthreads();
#pragma unroll
  for (int i = 0; i < 32; i += 8) {
    int n = n0 + ty + i, k = k0 + tx;
    out[(size_t)n * K + k] = (_Float16)tile[tx][ty + i];
  }
}

// bias padded with -1e30 so padded vocab columns contribute exp()=0
__global__ void pad_bias(const float* __restrict__ bo, float* __restrict__ bop,
                         int N, int Npad) {
  int v = blockIdx.x * blockDim.x + threadIdx.x;
  if (v < Npad) bop[v] = (v < N) ? bo[v] : -1.0e30f;
}

// Build E[8192][1024] f16: 4-token context window, embed row 0 forced to zero.
__global__ void gather_embed(const int* __restrict__ text,
                             const float* __restrict__ embed,
                             _Float16* __restrict__ E) {
  size_t idx = (size_t)blockIdx.x * blockDim.x + threadIdx.x; // TOKENS*1024
  int c = (int)(idx & 1023);
  int t = (int)(idx >> 10);
  int b = t >> 9, s = t & 511;
  int j = c >> 8;            // which of the 4 context slots
  int pos = s + j - 4;       // tokens s-4..s-1
  int id = (pos < 0) ? 0 : text[b * 512 + pos];
  float v = (id == 0) ? 0.0f : embed[(size_t)id * EMB + (c & 255)];
  E[idx] = (_Float16)v;
}

// ---------- GEMM + bias + relu (layers 1 & 2): C = relu(A @ B + bias) ----------
// A[M][K] f16 row-major, Bt[N][K] f16, C[M][N] f16.
// Block: 256 threads = 8 waves, tile 128(M) x 64(N); wave tile 32x32 (2x2 WMMA).
__global__ void gemm_bias_relu(const _Float16* __restrict__ A,
                               const _Float16* __restrict__ Bt,
                               const float* __restrict__ bias,
                               _Float16* __restrict__ C,
                               int M, int N, int K) {
  int lane = threadIdx.x & 31;
  int w = threadIdx.x >> 5;
  int wm = w >> 1, wn = w & 1;
  int m0 = blockIdx.x * 128 + wm * 32;
  int n0 = blockIdx.y * 64 + wn * 32;

  float8 acc[2][2] = {};
  for (int k0 = 0; k0 < K; k0 += 32) {
    half16 a0 = load_a_frag(A, K, m0, k0);
    half16 a1 = load_a_frag(A, K, m0 + 16, k0);
    half16 b0 = load_b_frag(Bt, K, n0, k0);
    half16 b1 = load_b_frag(Bt, K, n0 + 16, k0);
    acc[0][0] = wmma_f16(a0, b0, acc[0][0]);
    acc[0][1] = wmma_f16(a0, b1, acc[0][1]);
    acc[1][0] = wmma_f16(a1, b0, acc[1][0]);
    acc[1][1] = wmma_f16(a1, b1, acc[1][1]);
  }
  int hi = lane >> 4, ln = lane & 15;
#pragma unroll
  for (int i = 0; i < 2; ++i)
#pragma unroll
    for (int j = 0; j < 2; ++j) {
      int col = n0 + j * 16 + ln;
      float bv = bias[col];
#pragma unroll
      for (int r = 0; r < 8; ++r) {
        int row = m0 + i * 16 + r + 8 * hi;
        float v = fmaxf(acc[i][j][r] + bv, 0.0f);
        C[(size_t)row * N + col] = (_Float16)v;
      }
    }
}

// ---------- Fused output GEMM + online log-softmax partials ----------
// grid = (TOKENS/64, NCHUNK); block = 128 threads (4 waves, 16 rows each).
// 64 rows of h2 staged into LDS via CDNA5 async global->LDS copies; each wave
// processes TWO 16-col B tiles per iteration so the A fragment and the
// online-softmax cross-lane reductions are amortized over 2x the columns.
// Logits never hit memory - only per-(token,chunk) (max, sumexp) partials.
__global__ void logits_softmax_partial(const _Float16* __restrict__ H,
                                       const _Float16* __restrict__ Wot, // [VPAD][HID]
                                       const float* __restrict__ bop,    // [VPAD]
                                       float* __restrict__ pm,
                                       float* __restrict__ ps) {
  extern __shared__ _Float16 As[]; // 64 rows * 1024 cols f16 = 128 KB
  int tid = threadIdx.x;
  int mbase = blockIdx.x * 64;

  // stage 64 rows of H into LDS with async tensor-path copies (16B per lane)
  {
    unsigned lbase = (unsigned)(uintptr_t)As;              // LDS byte offset
    const _Float16* gsrc = H + (size_t)mbase * HID;
#pragma unroll 4
    for (int it = 0; it < 64; ++it) {
      int e = (it * 128 + tid) * 8;                        // element index
      unsigned lds = lbase + (unsigned)(e * 2);
      unsigned long long ga = (unsigned long long)(uintptr_t)(gsrc + e);
      asm volatile("global_load_async_to_lds_b128 %0, %1, off"
                   :: "v"(lds), "v"(ga) : "memory");
    }
    asm volatile("s_wait_asynccnt 0x0" ::: "memory");
  }
  __syncthreads();

  int lane = tid & 31, w = tid >> 5;
  int hi = lane >> 4, ln = lane & 15;

  float m_run[8], s_run[8];
#pragma unroll
  for (int r = 0; r < 8; ++r) { m_run[r] = -3.0e38f; s_run[r] = 0.0f; }

  int nstart = blockIdx.y * 4096;
  for (int nt = nstart; nt < nstart + 4096; nt += 32) {
    if (nt + 32 < nstart + 4096) {
      __builtin_prefetch(Wot + (size_t)(nt + 32 + ln) * HID, 0, 1); // global_prefetch_b8
      __builtin_prefetch(Wot + (size_t)(nt + 48 + ln) * HID, 0, 1);
    }
    float8 acc0 = {}, acc1 = {};
#pragma unroll 8
    for (int k0 = 0; k0 < HID; k0 += 32) {
      half16 a  = load_a_frag(As, HID, w * 16, k0);   // reused for both tiles
      half16 b0 = load_b_frag(Wot, HID, nt, k0);
      half16 b1 = load_b_frag(Wot, HID, nt + 16, k0);
      acc0 = wmma_f16(a, b0, acc0);
      acc1 = wmma_f16(a, b1, acc1);
    }
    float bv0 = bop[nt + ln];
    float bv1 = bop[nt + 16 + ln];
#pragma unroll
    for (int r = 0; r < 8; ++r) {
      float v0 = acc0[r] + bv0;
      float v1 = acc1[r] + bv1;
      // per-row max over the 16 lanes holding this row (xor<=8 stays in half)
      float mx = fmaxf(v0, v1);
      for (int off = 8; off; off >>= 1) mx = fmaxf(mx, __shfl_xor(mx, off, 32));
      float nm = fmaxf(m_run[r], mx);
      float p = __expf(v0 - nm) + __expf(v1 - nm);
      for (int off = 8; off; off >>= 1) p += __shfl_xor(p, off, 32);
      s_run[r] = s_run[r] * __expf(m_run[r] - nm) + p;
      m_run[r] = nm;
    }
  }
  if (ln == 0) {
#pragma unroll
    for (int r = 0; r < 8; ++r) {
      int row = mbase + w * 16 + r + 8 * hi;
      pm[row * NCHUNK + blockIdx.y] = m_run[r];
      ps[row * NCHUNK + blockIdx.y] = s_run[r];
    }
  }
}

// ---------- logit at target index: one wave per token ----------
__global__ void target_logit(const _Float16* __restrict__ H,
                             const _Float16* __restrict__ Wot,
                             const float* __restrict__ bo,
                             const int* __restrict__ target,
                             float* __restrict__ tl) {
  int w = threadIdx.x >> 5, lane = threadIdx.x & 31;
  int t = blockIdx.x * 8 + w;
  int tg = target[t];
  const _Float16* h = H + (size_t)t * HID;
  const _Float16* wv = Wot + (size_t)tg * HID;
  float sum = 0.0f;
#pragma unroll 8
  for (int kk = 0; kk < 32; ++kk) {
    int k = kk * 32 + lane;
    sum += (float)h[k] * (float)wv[k];
  }
  for (int off = 16; off; off >>= 1) sum += __shfl_xor(sum, off, 32);
  if (lane == 0) tl[t] = sum + bo[tg];
}

// ---------- combine partials, NLL, masked per-step mean, final mean ----------
__global__ void final_loss(const float* __restrict__ pm, const float* __restrict__ ps,
                           const float* __restrict__ tl, const int* __restrict__ target,
                           float* __restrict__ out) {
  __shared__ float red[512];
  int s = threadIdx.x; // 512 steps
  float acc = 0.0f, cnt = 0.0f;
  for (int b = 0; b < 16; ++b) {
    int t = b * 512 + s;
    float M = -3.0e38f;
#pragma unroll
    for (int c = 0; c < NCHUNK; ++c) M = fmaxf(M, pm[t * NCHUNK + c]);
    float S = 0.0f;
#pragma unroll
    for (int c = 0; c < NCHUNK; ++c) S += ps[t * NCHUNK + c] * __expf(pm[t * NCHUNK + c] - M);
    float lse = M + logf(S);
    float nll = lse - tl[t];
    if (target[t] != 0) { acc += nll; cnt += 1.0f; }
  }
  red[s] = acc / fmaxf(cnt, 1.0f);
  __syncthreads();
  for (int stride = 256; stride; stride >>= 1) {
    if (s < stride) red[s] += red[s + stride];
    __syncthreads();
  }
  if (s == 0) out[0] = red[0] / 512.0f;
}

// ---------- host launcher ----------

extern "C" void kernel_launch(void* const* d_in, const int* in_sizes, int n_in,
                              void* d_out, int out_size, void* d_ws, size_t ws_size,
                              hipStream_t stream) {
  const int*   text   = (const int*)d_in[0];
  const int*   target = (const int*)d_in[1];
  const float* embed  = (const float*)d_in[2];
  const float* W1     = (const float*)d_in[3];
  const float* b1     = (const float*)d_in[4];
  const float* W2     = (const float*)d_in[5];
  const float* b2     = (const float*)d_in[6];
  const float* Wo     = (const float*)d_in[7];
  const float* bo     = (const float*)d_in[8];
  float* out = (float*)d_out;

  char* ws = (char*)d_ws;
  const size_t OFF_E   = 0;                          // 16 MB  (E, later reused as h2)
  const size_t OFF_H1  = OFF_E   + (size_t)TOKENS * HID * 2;   // 16 MB
  const size_t OFF_W1T = OFF_H1  + (size_t)TOKENS * HID * 2;   // 2 MB
  const size_t OFF_W2T = OFF_W1T + (size_t)HID * HID * 2;      // 2 MB
  const size_t OFF_WOT = OFF_W2T + (size_t)HID * HID * 2;      // 64 MB
  const size_t OFF_BOP = OFF_WOT + (size_t)VPAD * HID * 2;     // 128 KB
  const size_t OFF_PM  = OFF_BOP + (size_t)VPAD * 4;           // 256 KB
  const size_t OFF_PS  = OFF_PM  + (size_t)TOKENS * NCHUNK * 4;// 256 KB
  const size_t OFF_TL  = OFF_PS  + (size_t)TOKENS * NCHUNK * 4;// 32 KB

  _Float16* E   = (_Float16*)(ws + OFF_E);
  _Float16* H1  = (_Float16*)(ws + OFF_H1);
  _Float16* H2  = (_Float16*)(ws + OFF_E);   // reuse E buffer for h2
  _Float16* W1t = (_Float16*)(ws + OFF_W1T);
  _Float16* W2t = (_Float16*)(ws + OFF_W2T);
  _Float16* WOt = (_Float16*)(ws + OFF_WOT);
  float* bop = (float*)(ws + OFF_BOP);
  float* pm  = (float*)(ws + OFF_PM);
  float* ps  = (float*)(ws + OFF_PS);
  float* tl  = (float*)(ws + OFF_TL);

  // 1) weight conversion/transpose to f16 [N][K] (tiled, coalesced both sides)
  dim3 tb(32, 8);
  conv_transpose_f16<<<dim3(HID / 32, HID / 32), tb, 0, stream>>>(W1, W1t, HID, HID, HID);
  conv_transpose_f16<<<dim3(HID / 32, HID / 32), tb, 0, stream>>>(W2, W2t, HID, HID, HID);
  conv_transpose_f16<<<dim3(VPAD / 32, HID / 32), tb, 0, stream>>>(Wo, WOt, HID, VOCAB, VPAD);
  pad_bias<<<(VPAD + 255) / 256, 256, 0, stream>>>(bo, bop, VOCAB, VPAD);

  // 2) embedding gather -> E f16
  gather_embed<<<((size_t)TOKENS * HID + 255) / 256, 256, 0, stream>>>(text, embed, E);

  // 3) h1 = relu(E @ W1 + b1); h2 = relu(h1 @ W2 + b2)
  dim3 g1(TOKENS / 128, HID / 64);
  gemm_bias_relu<<<g1, 256, 0, stream>>>(E, W1t, b1, H1, TOKENS, HID, HID);
  gemm_bias_relu<<<g1, 256, 0, stream>>>(H1, W2t, b2, H2, TOKENS, HID, HID);

  // 4) fused output GEMM + online softmax partials (logits never materialized)
  dim3 g2(TOKENS / 64, NCHUNK);
  logits_softmax_partial<<<g2, 128, 64 * HID * sizeof(_Float16), stream>>>(H2, WOt, bop, pm, ps);

  // 5) target logits
  target_logit<<<TOKENS / 8, 256, 0, stream>>>(H2, WOt, bo, target, tl);

  // 6) combine + masked mean -> scalar loss
  final_loss<<<1, 512, 0, stream>>>(pm, ps, tl, target, out);
}
